// PointNetSetAbstractionMsg_77068893160405
// MI455X (gfx1250) — matrix operand order repaired and checked by reading, not compiled, not run on hardware
//
#include <hip/hip_runtime.h>
#include <cstdint>

typedef __attribute__((ext_vector_type(16))) _Float16 v16h;
typedef __attribute__((ext_vector_type(8)))  float    v8f;
typedef unsigned int uint;

#define B_   16
#define N_   4096
#define CIN_ 6
#define S_   1024
#define EPS_ 1e-3f

// ---- workspace layout (bytes) ----
#define WS_FPSIDX  0u                         // 16384 int
#define WS_GIDX    65536u                     // 16*1024*128 int = 8 MB
#define WS_WPACK   8454144u                   // 9 slots * 32768 B packed f16 weights
#define WS_STATS   (8454144u + 9u*32768u)     // 9 slots * 512 floats: sum|sumsq|a|b @ +0/+128/+256/+384

// ============================================================ FPS
__global__ __launch_bounds__(256) void fps_kernel(const float* __restrict__ xyz,
                                                  int* __restrict__ fps_idx,
                                                  float* out_newxyz) {
  __shared__ float dist[N_];
  __shared__ float rv[256];
  __shared__ int   ri[256];
  __shared__ int   cur;
  const int b = blockIdx.x, tid = threadIdx.x;
  for (int i = tid; i < N_; i += 256) dist[i] = 1e10f;
  if (tid == 0) cur = 0;
  __syncthreads();
  for (int s = 0; s < S_; ++s) {
    int far = cur;
    const float* c = xyz + (size_t)(b * N_ + far) * 3;
    float cx = c[0], cy = c[1], cz = c[2];
    if (tid == 0) {
      fps_idx[b * S_ + s] = far;
      float* o = out_newxyz + (size_t)(b * S_ + s) * 3;
      o[0] = cx; o[1] = cy; o[2] = cz;
    }
    float bv = -1.f; int bi = 0;
    for (int i = tid; i < N_; i += 256) {
      const float* p = xyz + (size_t)(b * N_ + i) * 3;
      float dx = p[0] - cx, dy = p[1] - cy, dz = p[2] - cz;
      float d = dx * dx + dy * dy + dz * dz;
      float nd = fminf(dist[i], d);
      dist[i] = nd;
      if (nd > bv) { bv = nd; bi = i; }
    }
    rv[tid] = bv; ri[tid] = bi;
    __syncthreads();
    for (int off = 128; off > 0; off >>= 1) {
      if (tid < off) {
        float ov = rv[tid + off]; int oi = ri[tid + off];
        if (ov > rv[tid] || (ov == rv[tid] && oi < ri[tid])) { rv[tid] = ov; ri[tid] = oi; }
      }
      __syncthreads();
    }
    if (tid == 0) cur = ri[0];
    __syncthreads();
  }
}

// ============================================================ query_ball
// K smallest in-ball indices (ascending scan == top_k(-idx) semantics), pad with first hit.
template<int K>
__global__ __launch_bounds__(256) void qball_kernel(const float* __restrict__ xyz,
                                                    const float* newxyz,
                                                    int* __restrict__ gidx, float r2) {
  const int w = blockIdx.x * 8 + (threadIdx.x >> 5);
  const int lane = threadIdx.x & 31;
  const int b = w >> 10;
  const float* c = newxyz + (size_t)w * 3;
  const float cx = c[0], cy = c[1], cz = c[2];
  const float* base = xyz + (size_t)b * N_ * 3;
  int cnt = 0, first = 0; bool havef = false;
  for (int j0 = 0; j0 < N_ && cnt < K; j0 += 32) {
    int i = j0 + lane;
    const float* p = base + (size_t)i * 3;
    float dx = p[0] - cx, dy = p[1] - cy, dz = p[2] - cz;
    bool in = (dx * dx + dy * dy + dz * dz) <= r2;
    uint m = (uint)__ballot(in);
    if (!havef && m) { first = j0 + __builtin_ctz(m); havef = true; }
    if (in) {
      int slot = cnt + __popc(m & ((1u << lane) - 1u));
      if (slot < K) gidx[(size_t)w * K + slot] = i;
    }
    cnt += __popc(m);
  }
  int nf = cnt < K ? cnt : K;
  for (int t = nf + lane; t < K; t += 32) gidx[(size_t)w * K + t] = first;
}

// ============================================================ weight packing (f32 -> f16 B-fragment layout)
// dst[(((nt*KT+kt)*32)+lane)*8 + v] = packed halves for K = kt*32 + g*8 + (v<4 ? 2v : 16+2(v-4)) (+1), col = nt*16 + (lane&15)
__global__ void pack_w_kernel(const float* __restrict__ W, int cin, int f, int KT,
                              uint* __restrict__ dst) {
  int NT = f >> 4;
  int total = NT * KT * 256;
  int gid = blockIdx.x * 256 + threadIdx.x;
  if (gid >= total) return;
  int v = gid & 7;
  int lane = (gid >> 3) & 31;
  int tile = gid >> 8;               // nt*KT + kt
  int kt = tile % KT;
  int g = lane >> 4;
  int col = ((tile / KT) * 16) + (lane & 15);
  int k0 = kt * 32 + g * 8 + ((v < 4) ? 2 * v : 16 + 2 * (v - 4));
  float f0 = (k0     < cin) ? W[(size_t)k0 * f + col]       : 0.f;
  float f1 = (k0 + 1 < cin) ? W[(size_t)(k0 + 1) * f + col] : 0.f;
  _Float16 h0 = (_Float16)f0, h1 = (_Float16)f1;
  uint u = ((uint)*(unsigned short*)&h1 << 16) | (uint)*(unsigned short*)&h0;
  dst[gid] = u;
}

__global__ void zero_stats_kernel(float* s, int n) {
  int i = blockIdx.x * 256 + threadIdx.x;
  if (i < n) s[i] = 0.f;
}

// mean/var -> folded scale/shift:  xn = a*y + b
__global__ void finalize_stats_kernel(float* slot, const float* __restrict__ gamma,
                                      const float* __restrict__ beta, int f, float invc) {
  int i = threadIdx.x;
  if (i < f) {
    float mean = slot[i] * invc;
    float var  = slot[128 + i] * invc - mean * mean;
    float a = gamma[i] * rsqrtf(var + EPS_);
    slot[256 + i] = a;
    slot[384 + i] = beta[i] - mean * a;
  }
}

// ============================================================ one MLP layer on a 16-row tile (per wave)
// MODE 0: relu -> normalize -> store f16 back to X (stride POUT)
// MODE 1: relu -> accumulate per-lane sum/sumsq
// MODE 2: relu -> normalize -> running max in registers
template<int KT, int NT, int PIN, int POUT, int MODE>
__device__ __forceinline__ void layer_step(_Float16* X, const uint* __restrict__ Wp,
                                           const float* __restrict__ bias,
                                           const float* __restrict__ cA,
                                           const float* __restrict__ cB,
                                           float* ssum, float* ssq, float* mx, int lane) {
  v8f acc[NT];
#pragma unroll
  for (int nt = 0; nt < NT; ++nt) acc[nt] = {};
  const int r = lane & 15, g = lane >> 4;
  const _Float16* Xr = X + r * PIN;
#pragma unroll
  for (int kt = 0; kt < KT; ++kt) {
    union { uint u[8]; v16h h; } a;
#pragma unroll
    for (int v = 0; v < 8; ++v) {
      int k0 = kt * 32 + g * 8 + ((v < 4) ? 2 * v : 16 + 2 * (v - 4));
      a.u[v] = *(const uint*)(Xr + k0);
    }
#pragma unroll
    for (int nt = 0; nt < NT; ++nt) {
      union { uint4 q[2]; v16h h; } bf;
      const uint4* bp = (const uint4*)(Wp + ((size_t)(nt * KT + kt) * 32 + lane) * 8);
      bf.q[0] = bp[0]; bf.q[1] = bp[1];
      acc[nt] = __builtin_amdgcn_wmma_f32_16x16x32_f16(false, a.h, false, bf.h,
                                                       (short)0, acc[nt], false, false);
    }
  }
  __syncthreads();   // all reads of X done before (possible) overwrite
#pragma unroll
  for (int nt = 0; nt < NT; ++nt) {
    int col = nt * 16 + (lane & 15);
    float bs = bias[col];
    float av = 0.f, bv = 0.f;
    if constexpr (MODE != 1) { av = cA[col]; bv = cB[col]; }
#pragma unroll
    for (int i = 0; i < 8; ++i) {
      float y = acc[nt][i] + bs;
      y = y > 0.f ? y : 0.f;
      if constexpr (MODE == 1) { ssum[nt] += y; ssq[nt] += y * y; }
      else {
        float xn = av * y + bv;
        if constexpr (MODE == 0) {
          int m = i + 8 * g;                       // D layout: row = i + 8*g
          X[m * POUT + col] = (_Float16)xn;
        } else {
          float cm = mx[nt * 8 + i];
          mx[nt * 8 + i] = xn > cm ? xn : cm;
        }
      }
    }
  }
  __syncthreads();   // stores visible before next layer reads
}

// ============================================================ fused pass kernel (1 wave = 1 center)
template<int K, int C1, int C2, int C3, int OFF, int STAGE, int PMAX>
__global__ __launch_bounds__(128) void pass_kernel(
    const float* __restrict__ xyz, const float* __restrict__ points,
    const float* newxyz, const int* __restrict__ gidx,
    const uint* __restrict__ w1, const uint* __restrict__ w2, const uint* __restrict__ w3,
    const float* __restrict__ b1, const float* __restrict__ b2, const float* __restrict__ b3,
    float* st1, float* st2, float* st3, float* out) {
  constexpr int NT1 = C1 / 16, NT2 = C2 / 16, NT3 = C3 / 16;
  constexpr int KT1 = 1, KT2 = C1 / 32, KT3 = C2 / 32;
  constexpr int NTS = (STAGE == 1) ? NT1 : (STAGE == 2) ? NT2 : NT3;
  __shared__ _Float16 shX[4][16 * PMAX];
  const int wid = threadIdx.x >> 5, lane = threadIdx.x & 31;
  const int center = blockIdx.x * 4 + wid;
  _Float16* X = &shX[wid][0];
  const int b = center >> 10;
  const float* c = newxyz + (size_t)center * 3;
  const float cx = c[0], cy = c[1], cz = c[2];
  float ssum[NTS], ssq[NTS];
#pragma unroll
  for (int i = 0; i < NTS; ++i) { ssum[i] = 0.f; ssq[i] = 0.f; }
  float mx[(STAGE == 4) ? (NT3 * 8) : 1];
#pragma unroll
  for (int i = 0; i < ((STAGE == 4) ? (NT3 * 8) : 1); ++i) mx[i] = -3.4e38f;
  const int r = lane & 15, g = lane >> 4;

  for (int mt = 0; mt < K / 16; ++mt) {
    // gather X0 tile: 16 rows x 32 cols (cols 0-5 points, 6-8 rel xyz, 9-31 zero pad)
    int gi = gidx[(size_t)center * K + mt * 16 + r];
    const float* pp = points + (size_t)(b * N_ + gi) * CIN_;
    const float* px = xyz + (size_t)(b * N_ + gi) * 3;
#pragma unroll
    for (int j = 0; j < 16; ++j) {
      int cc = g * 16 + j;
      float v;
      if (cc < 6)      v = pp[cc];
      else if (cc < 9) v = px[cc - 6] - (cc == 6 ? cx : (cc == 7 ? cy : cz));
      else             v = 0.f;
      X[r * 32 + cc] = (_Float16)v;
    }
    __syncthreads();
    if constexpr (STAGE == 1) {
      layer_step<KT1, NT1, 32, 32, 1>(X, w1, b1, nullptr, nullptr, ssum, ssq, nullptr, lane);
    } else {
      layer_step<KT1, NT1, 32, C1, 0>(X, w1, b1, st1 + 256, st1 + 384, nullptr, nullptr, nullptr, lane);
      if constexpr (STAGE == 2) {
        layer_step<KT2, NT2, C1, C1, 1>(X, w2, b2, nullptr, nullptr, ssum, ssq, nullptr, lane);
      } else {
        layer_step<KT2, NT2, C1, C2, 0>(X, w2, b2, st2 + 256, st2 + 384, nullptr, nullptr, nullptr, lane);
        if constexpr (STAGE == 3) {
          layer_step<KT3, NT3, C2, C2, 1>(X, w3, b3, nullptr, nullptr, ssum, ssq, nullptr, lane);
        } else {
          layer_step<KT3, NT3, C2, C2, 2>(X, w3, b3, st3 + 256, st3 + 384, nullptr, nullptr, mx, lane);
        }
      }
    }
  }

  if constexpr (STAGE <= 3) {
    float* st = (STAGE == 1) ? st1 : (STAGE == 2) ? st2 : st3;
#pragma unroll
    for (int nt = 0; nt < NTS; ++nt) {
      float s0 = ssum[nt] + __shfl_xor(ssum[nt], 16, 32);
      float s1 = ssq[nt]  + __shfl_xor(ssq[nt], 16, 32);
      if (lane < 16) {
        atomicAdd(st + nt * 16 + lane, s0);
        atomicAdd(st + 128 + nt * 16 + lane, s1);
      }
    }
  } else {
#pragma unroll
    for (int nt = 0; nt < NT3; ++nt) {
      float m = -3.4e38f;
#pragma unroll
      for (int i = 0; i < 8; ++i) m = fmaxf(m, mx[nt * 8 + i]);
      float o = fmaxf(m, __shfl_xor(m, 16, 32));
      if (lane < 16) {
        int col = nt * 16 + lane;
        out[(size_t)(B_ * S_ * 3) + (size_t)center * 320 + OFF + col] = o;
      }
    }
  }
}

// ============================================================ host
extern "C" void kernel_launch(void* const* d_in, const int* in_sizes, int n_in,
                              void* d_out, int out_size, void* d_ws, size_t ws_size,
                              hipStream_t stream) {
  const float* xyz    = (const float*)d_in[0];
  const float* points = (const float*)d_in[1];
  auto P = [&](int scale, int layer, int which) -> const float* {
    return (const float*)d_in[2 + (scale * 3 + layer) * 4 + which];   // W,b,gamma,beta
  };
  char* ws = (char*)d_ws;
  int*  fpsidx = (int*)(ws + WS_FPSIDX);
  int*  gidx   = (int*)(ws + WS_GIDX);
  uint*  wslot[9];
  float* sslot[9];
  for (int i = 0; i < 9; ++i) {
    wslot[i] = (uint*)(ws + WS_WPACK + (size_t)i * 32768u);
    sslot[i] = (float*)(ws + WS_STATS + (size_t)i * 2048u);
  }
  float* out = (float*)d_out;
  float* newxyz = out;  // first B*S*3 floats of the output tuple

  zero_stats_kernel<<<(9 * 512 + 255) / 256, 256, 0, stream>>>((float*)(ws + WS_STATS), 9 * 512);
  fps_kernel<<<16, 256, 0, stream>>>(xyz, fpsidx, newxyz);

  const int cins[9] = {9, 32, 32,   9, 64, 64,   9, 64, 96};
  const int fs[9]   = {32, 32, 64,  64, 64, 128, 64, 96, 128};
  for (int i = 0; i < 9; ++i) {
    int scale = i / 3, layer = i % 3;
    int KT = (cins[i] + 31) / 32;
    int total = (fs[i] / 16) * KT * 256;
    pack_w_kernel<<<(total + 255) / 256, 256, 0, stream>>>(P(scale, layer, 0), cins[i], fs[i], KT, wslot[i]);
  }

  // ---- scale 0: K=16, 9->32->32->64, radius 0.1, out offset 0
  {
    qball_kernel<16><<<2048, 256, 0, stream>>>(xyz, newxyz, gidx, 0.01f);
    float invc = 1.f / (float)(B_ * S_ * 16);
    pass_kernel<16, 32, 32, 64, 0, 1, 32><<<4096, 128, 0, stream>>>(xyz, points, newxyz, gidx,
        wslot[0], wslot[1], wslot[2], P(0,0,1), P(0,1,1), P(0,2,1), sslot[0], sslot[1], sslot[2], out);
    finalize_stats_kernel<<<1, 128, 0, stream>>>(sslot[0], P(0,0,2), P(0,0,3), 32, invc);
    pass_kernel<16, 32, 32, 64, 0, 2, 32><<<4096, 128, 0, stream>>>(xyz, points, newxyz, gidx,
        wslot[0], wslot[1], wslot[2], P(0,0,1), P(0,1,1), P(0,2,1), sslot[0], sslot[1], sslot[2], out);
    finalize_stats_kernel<<<1, 128, 0, stream>>>(sslot[1], P(0,1,2), P(0,1,3), 32, invc);
    pass_kernel<16, 32, 32, 64, 0, 3, 32><<<4096, 128, 0, stream>>>(xyz, points, newxyz, gidx,
        wslot[0], wslot[1], wslot[2], P(0,0,1), P(0,1,1), P(0,2,1), sslot[0], sslot[1], sslot[2], out);
    finalize_stats_kernel<<<1, 128, 0, stream>>>(sslot[2], P(0,2,2), P(0,2,3), 64, invc);
    pass_kernel<16, 32, 32, 64, 0, 4, 32><<<4096, 128, 0, stream>>>(xyz, points, newxyz, gidx,
        wslot[0], wslot[1], wslot[2], P(0,0,1), P(0,1,1), P(0,2,1), sslot[0], sslot[1], sslot[2], out);
  }
  // ---- scale 1: K=32, 9->64->64->128, radius 0.2, out offset 64
  {
    qball_kernel<32><<<2048, 256, 0, stream>>>(xyz, newxyz, gidx, 0.04f);
    float invc = 1.f / (float)(B_ * S_ * 32);
    pass_kernel<32, 64, 64, 128, 64, 1, 64><<<4096, 128, 0, stream>>>(xyz, points, newxyz, gidx,
        wslot[3], wslot[4], wslot[5], P(1,0,1), P(1,1,1), P(1,2,1), sslot[3], sslot[4], sslot[5], out);
    finalize_stats_kernel<<<1, 128, 0, stream>>>(sslot[3], P(1,0,2), P(1,0,3), 64, invc);
    pass_kernel<32, 64, 64, 128, 64, 2, 64><<<4096, 128, 0, stream>>>(xyz, points, newxyz, gidx,
        wslot[3], wslot[4], wslot[5], P(1,0,1), P(1,1,1), P(1,2,1), sslot[3], sslot[4], sslot[5], out);
    finalize_stats_kernel<<<1, 128, 0, stream>>>(sslot[4], P(1,1,2), P(1,1,3), 64, invc);
    pass_kernel<32, 64, 64, 128, 64, 3, 64><<<4096, 128, 0, stream>>>(xyz, points, newxyz, gidx,
        wslot[3], wslot[4], wslot[5], P(1,0,1), P(1,1,1), P(1,2,1), sslot[3], sslot[4], sslot[5], out);
    finalize_stats_kernel<<<1, 128, 0, stream>>>(sslot[5], P(1,2,2), P(1,2,3), 128, invc);
    pass_kernel<32, 64, 64, 128, 64, 4, 64><<<4096, 128, 0, stream>>>(xyz, points, newxyz, gidx,
        wslot[3], wslot[4], wslot[5], P(1,0,1), P(1,1,1), P(1,2,1), sslot[3], sslot[4], sslot[5], out);
  }
  // ---- scale 2: K=128, 9->64->96->128, radius 0.4, out offset 192
  {
    qball_kernel<128><<<2048, 256, 0, stream>>>(xyz, newxyz, gidx, 0.16f);
    float invc = 1.f / (float)(B_ * S_ * 128);
    pass_kernel<128, 64, 96, 128, 192, 1, 96><<<4096, 128, 0, stream>>>(xyz, points, newxyz, gidx,
        wslot[6], wslot[7], wslot[8], P(2,0,1), P(2,1,1), P(2,2,1), sslot[6], sslot[7], sslot[8], out);
    finalize_stats_kernel<<<1, 128, 0, stream>>>(sslot[6], P(2,0,2), P(2,0,3), 64, invc);
    pass_kernel<128, 64, 96, 128, 192, 2, 96><<<4096, 128, 0, stream>>>(xyz, points, newxyz, gidx,
        wslot[6], wslot[7], wslot[8], P(2,0,1), P(2,1,1), P(2,2,1), sslot[6], sslot[7], sslot[8], out);
    finalize_stats_kernel<<<1, 128, 0, stream>>>(sslot[7], P(2,1,2), P(2,1,3), 96, invc);
    pass_kernel<128, 64, 96, 128, 192, 3, 96><<<4096, 128, 0, stream>>>(xyz, points, newxyz, gidx,
        wslot[6], wslot[7], wslot[8], P(2,0,1), P(2,1,1), P(2,2,1), sslot[6], sslot[7], sslot[8], out);
    finalize_stats_kernel<<<1, 128, 0, stream>>>(sslot[8], P(2,2,2), P(2,2,3), 128, invc);
    pass_kernel<128, 64, 96, 128, 192, 4, 96><<<4096, 128, 0, stream>>>(xyz, points, newxyz, gidx,
        wslot[6], wslot[7], wslot[8], P(2,0,1), P(2,1,1), P(2,2,1), sslot[6], sslot[7], sslot[8], out);
  }
}
